// TransformerOCModel_11390253269524
// MI455X (gfx1250) — matrix operand order repaired and checked by reading, not compiled, not run on hardware
//
#include <hip/hip_runtime.h>
#include <hip/hip_bf16.h>
#include <math.h>

typedef _Float16 v16h __attribute__((ext_vector_type(16)));
typedef float    v8f  __attribute__((ext_vector_type(8)));
typedef _Float16 h2   __attribute__((ext_vector_type(2)));
typedef unsigned int v4u __attribute__((ext_vector_type(4)));
typedef int          v8i __attribute__((ext_vector_type(8)));
typedef int          v4i __attribute__((ext_vector_type(4)));

#define NNODES 8192
#define HID    256
#define NH     8
#define HD     32
#define KNBR   32
#define NLAY   4

__device__ __forceinline__ float wave_sum(float v) {
    #pragma unroll
    for (int off = 16; off >= 1; off >>= 1) v += __shfl_xor(v, off, 32);
    return v;
}
__device__ __forceinline__ float wave_max(float v) {
    #pragma unroll
    for (int off = 16; off >= 1; off >>= 1) v = fmaxf(v, __shfl_xor(v, off, 32));
    return v;
}

// ---------------------------------------------------------------------------
// TDM: 2D f16 tile (32 x tile_rows, row stride K elems) global -> LDS.
// Descriptor packing per CDNA5 ISA 8.3/8.4; groups 2/3 zero (2-D tile).
// clang-23 toolchain: 6-arg builtin (g0 v4u, g1 v8i, g2 v4i, g3 v4i, v8i, cpol).
// ---------------------------------------------------------------------------
__device__ __forceinline__ void tdm_load_tile_f16(const _Float16* gptr, unsigned lds_byte,
                                                  int tile_rows, int K)
{
    unsigned long long ga = (unsigned long long)(size_t)gptr;
    v4u g0;
    g0.x = 1u;                                           // count=1 (valid user D#)
    g0.y = lds_byte;                                     // lds_addr (bytes)
    g0.z = (unsigned)(ga & 0xFFFFFFFFull);               // global_addr[31:0]
    g0.w = (unsigned)((ga >> 32) & 0x01FFFFFFull)        // global_addr[56:32]
           | 0x80000000u;                                // type=2 ("image")
    v8i g1;
    g1[0] = 0x10000;                                     // data_size=1 (2 bytes); wg_mask=0
    g1[1] = (int)(((unsigned)K & 0xFFFFu) << 16);        // tensor_dim0[15:0] = K
    g1[2] = (int)((((unsigned)K >> 16) & 0xFFFFu));      // tensor_dim0[31:16]; tensor_dim1[15:0]=0
    g1[3] = (int)(0x10u | (32u << 16));                  // tensor_dim1=1<<20; tile_dim0=32
    g1[4] = tile_rows;                                   // tile_dim1 = rows; tile_dim2=0
    g1[5] = K;                                           // tensor_dim0_stride[31:0]
    g1[6] = 0;
    g1[7] = 0;
    v4i z4 = {0, 0, 0, 0};
    v8i z8 = {0, 0, 0, 0, 0, 0, 0, 0};
    __builtin_amdgcn_tensor_load_to_lds(g0, g1, z4, z4, z8, 0);
}

// ---------------------------------------------------------------------------
// Neighbor build: per thread = one query row; tile candidates through LDS,
// keep running top-32 smallest squared-dr with index tiebreak.
// ---------------------------------------------------------------------------
__global__ __launch_bounds__(256)
void neighbors_kernel(const float* __restrict__ x_raw, const int* __restrict__ batch,
                      int* __restrict__ idx_out, unsigned char* __restrict__ mask_out)
{
    __shared__ float se[256], sp[256];
    __shared__ int   sb[256];
    const int i = blockIdx.x * 256 + threadIdx.x;

    const float eta_i = x_raw[i * 16 + 1] * 5.24f - 2.62f;
    const float phi_i = x_raw[i * 16 + 2] * 6.2832f - 3.1416f;
    const int   b_i   = batch[i];

    float bd[KNBR];
    int   bi[KNBR];
    #pragma unroll
    for (int t = 0; t < KNBR; ++t) { bd[t] = 1e12f; bi[t] = 0x7fffffff; }
    float curmax = 1e12f; int curarg = 0; int curidx = 0x7fffffff;

    for (int j0 = 0; j0 < NNODES; j0 += 256) {
        __syncthreads();
        const int jj = j0 + threadIdx.x;
        se[threadIdx.x] = x_raw[jj * 16 + 1] * 5.24f - 2.62f;
        sp[threadIdx.x] = x_raw[jj * 16 + 2] * 6.2832f - 3.1416f;
        sb[threadIdx.x] = batch[jj];
        __syncthreads();
        for (int t = 0; t < 256; ++t) {
            const int j = j0 + t;
            const float de = eta_i - se[t];
            float dp = phi_i - sp[t];
            dp = dp - 6.2831853f * rintf(dp * 0.15915494f);  // wrap to [-pi,pi]
            float d2 = de * de + dp * dp;
            if (sb[t] != b_i) d2 = 1e12f;                    // cross-batch penalty
            if (d2 < curmax || (d2 == curmax && j < curidx)) {
                bd[curarg] = d2; bi[curarg] = j;
                curmax = bd[0]; curidx = bi[0]; curarg = 0;
                for (int t2 = 1; t2 < KNBR; ++t2) {
                    if (bd[t2] > curmax || (bd[t2] == curmax && bi[t2] > curidx)) {
                        curmax = bd[t2]; curidx = bi[t2]; curarg = t2;
                    }
                }
            }
        }
    }
    for (int a = 0; a < KNBR; ++a) {   // sort ascending (d2, idx)
        int best = a;
        for (int c = a + 1; c < KNBR; ++c)
            if (bd[c] < bd[best] || (bd[c] == bd[best] && bi[c] < bi[best])) best = c;
        float td = bd[a]; bd[a] = bd[best]; bd[best] = td;
        int   ti = bi[a]; bi[a] = bi[best]; bi[best] = ti;
    }
    bool any = false;
    #pragma unroll
    for (int t = 0; t < KNBR; ++t) any = any || (bd[t] <= 0.04f);  // 0.2^2
    if (!any) { bi[0] = i; bd[0] = 0.0f; }
    #pragma unroll
    for (int t = 0; t < KNBR; ++t) {
        idx_out[i * KNBR + t]  = bi[t];
        mask_out[i * KNBR + t] = (bd[t] <= 0.04f) ? 0 : 1;          // 1 = padding
    }
}

// ---------------------------------------------------------------------------
// Input projection: x = x_raw @ w_in + b_in   (8192x16 @ 16x256)
// ---------------------------------------------------------------------------
__global__ __launch_bounds__(256)
void inproj_kernel(const float* __restrict__ xr, const float* __restrict__ w,
                   const float* __restrict__ b, float* __restrict__ x)
{
    __shared__ float xs[16];
    const int n = blockIdx.x, c = threadIdx.x;
    if (threadIdx.x < 16) xs[threadIdx.x] = xr[n * 16 + threadIdx.x];
    __syncthreads();
    float s = b[c];
    #pragma unroll
    for (int i = 0; i < 16; ++i) s += xs[i] * w[i * HID + c];
    x[(size_t)n * HID + c] = s;
}

// ---------------------------------------------------------------------------
// Weight convert + transpose: Bt[p][k] = (f16) B[k][p]
// ---------------------------------------------------------------------------
__global__ __launch_bounds__(256)
void wcvt_kernel(const float* __restrict__ B, _Float16* __restrict__ Bt, int K, int P)
{
    const int p = blockIdx.x;
    for (int k = threadIdx.x; k < K; k += 256)
        Bt[(size_t)p * K + k] = (_Float16)B[(size_t)k * P + p];
}

// ---------------------------------------------------------------------------
// LayerNorm: one wave per node, 8 elems per lane; writes f32 + f16 copies.
// ---------------------------------------------------------------------------
__global__ __launch_bounds__(256)
void ln_kernel(const float* __restrict__ x, const float* __restrict__ g,
               const float* __restrict__ b, float* __restrict__ y,
               _Float16* __restrict__ y16)
{
    const int node = (blockIdx.x * 256 + threadIdx.x) >> 5;
    const int lane = threadIdx.x & 31;
    const float* xr = x + (size_t)node * HID;
    float vals[8];
    float s = 0.f;
    #pragma unroll
    for (int t = 0; t < 8; ++t) { vals[t] = xr[lane + t * 32]; s += vals[t]; }
    s = wave_sum(s);
    const float mu = s * (1.0f / HID);
    float vs = 0.f;
    #pragma unroll
    for (int t = 0; t < 8; ++t) { const float d = vals[t] - mu; vs += d * d; }
    vs = wave_sum(vs);
    const float inv = rsqrtf(vs * (1.0f / HID) + 1e-5f);
    #pragma unroll
    for (int t = 0; t < 8; ++t) {
        const int c = lane + t * 32;
        const float o = (vals[t] - mu) * inv * g[c] + b[c];
        y[(size_t)node * HID + c]   = o;
        y16[(size_t)node * HID + c] = (_Float16)o;
    }
}

// ---------------------------------------------------------------------------
// TDM + WMMA GEMM: C[M x P] = A16[M x K] @ Bt16[P x K]^T (+bias)(+res)(ReLU)
// block = 256 threads = 8 waves; block tile 128(M) x 64(N);
// wave = 16 rows x 4 WMMA N-tiles. Panels land in LDS via tensor_load_to_lds.
// ---------------------------------------------------------------------------
__global__ __launch_bounds__(256)
void gemm_wmma_tdm(const _Float16* __restrict__ A16, const _Float16* __restrict__ Bt16,
                   const float* __restrict__ bias, const float* __restrict__ residual,
                   float* __restrict__ C, _Float16* __restrict__ C16,
                   int M, int K, int P, int relu_flag)
{
    __shared__ _Float16 Alds[128 * 32];   // row-major [m][k]
    __shared__ _Float16 Blds[64 * 32];    // col-major [n][k]
    const int tid  = threadIdx.x;
    const int wave = tid >> 5, lane = tid & 31;
    const int bm = blockIdx.y * 128;
    const int bn = blockIdx.x * 64;
    const int hi = lane >> 4;
    const int mrow = (wave << 4) + (lane & 15);
    const int nsub = lane & 15;
    const unsigned aoff = (unsigned)(size_t)(void*)Alds;   // LDS byte offset (addr[31:0])
    const unsigned boff = (unsigned)(size_t)(void*)Blds;

    v8f acc[4] = {};
    for (int k0 = 0; k0 < K; k0 += 32) {
        if (wave == 0) {
            tdm_load_tile_f16(A16  + (size_t)bm * K + k0, aoff, 128, K);
            tdm_load_tile_f16(Bt16 + (size_t)bn * K + k0, boff, 64,  K);
            __builtin_amdgcn_s_wait_tensorcnt(0);
        }
        __syncthreads();

        v16h a;
        #pragma unroll
        for (int g = 0; g < 2; ++g)
            #pragma unroll
            for (int vv = 0; vv < 4; ++vv) {
                const int p  = g * 4 + vv;
                const int ka = g * 16 + hi * 8 + vv * 2;
                const h2 pa = *(const h2*)&Alds[mrow * 32 + ka];
                a[2 * p]     = pa.x;
                a[2 * p + 1] = pa.y;
            }
        #pragma unroll
        for (int nt = 0; nt < 4; ++nt) {
            v16h b;
            const int ncol = nt * 16 + nsub;
            #pragma unroll
            for (int vv = 0; vv < 8; ++vv) {
                const int kb = (vv >> 2) * 16 + hi * 8 + (vv & 3) * 2;
                const h2 pb = *(const h2*)&Blds[ncol * 32 + kb];
                b[2 * vv]     = pb.x;
                b[2 * vv + 1] = pb.y;
            }
            acc[nt] = __builtin_amdgcn_wmma_f32_16x16x32_f16(false, a, false, b,
                                                             (short)0, acc[nt], false, false);
        }
        __syncthreads();
    }

    #pragma unroll
    for (int nt = 0; nt < 4; ++nt) {
        const int col = bn + nt * 16 + nsub;
        const float bv = bias ? bias[col] : 0.f;
        #pragma unroll
        for (int vv = 0; vv < 8; ++vv) {
            const int row = bm + (wave << 4) + vv + hi * 8;
            float val = acc[nt][vv] + bv;
            if (residual) val += residual[(size_t)row * P + col];
            if (relu_flag) val = fmaxf(val, 0.f);
            C[(size_t)row * P + col] = val;
            if (C16) C16[(size_t)row * P + col] = (_Float16)val;
        }
    }
}

// ---------------------------------------------------------------------------
// Narrow f32-input WMMA GEMM (kept for P=16 head matmul).
// ---------------------------------------------------------------------------
__global__ __launch_bounds__(256)
void gemm_wmma16(const float* __restrict__ A, const float* __restrict__ B,
                 const float* __restrict__ bias, float* __restrict__ C,
                 int M, int K, int P)
{
    __shared__ _Float16 Alds[128 * 32];
    __shared__ _Float16 Blds[16 * 32];
    const int tid  = threadIdx.x;
    const int wave = tid >> 5, lane = tid & 31;
    const int bm = blockIdx.y * 128;
    const int bn = blockIdx.x * 16;
    const int hi = lane >> 4;
    const int mrow = (wave << 4) + (lane & 15);
    const int ncol = lane & 15;

    v8f acc = {};
    for (int k0 = 0; k0 < K; k0 += 32) {
        for (int i = tid; i < 128 * 32; i += 256) {
            const int r = i >> 5, c = i & 31;
            Alds[i] = (_Float16)A[(size_t)(bm + r) * K + (k0 + c)];
        }
        for (int i = tid; i < 16 * 32; i += 256) {
            const int n = i >> 5, c = i & 31;
            Blds[i] = (_Float16)B[(size_t)(k0 + c) * P + (bn + n)];
        }
        __syncthreads();
        v16h a, b;
        #pragma unroll
        for (int g = 0; g < 2; ++g)
            #pragma unroll
            for (int vv = 0; vv < 4; ++vv) {
                const int p  = g * 4 + vv;
                const int ka = g * 16 + hi * 8 + vv * 2;
                const h2 pa = *(const h2*)&Alds[mrow * 32 + ka];
                a[2 * p]     = pa.x;
                a[2 * p + 1] = pa.y;
            }
        #pragma unroll
        for (int vv = 0; vv < 8; ++vv) {
            const int kb = (vv >> 2) * 16 + hi * 8 + (vv & 3) * 2;
            const h2 pb = *(const h2*)&Blds[ncol * 32 + kb];
            b[2 * vv]     = pb.x;
            b[2 * vv + 1] = pb.y;
        }
        acc = __builtin_amdgcn_wmma_f32_16x16x32_f16(false, a, false, b,
                                                     (short)0, acc, false, false);
        __syncthreads();
    }
    const int col = bn + ncol;
    const float bv = bias ? bias[col] : 0.f;
    #pragma unroll
    for (int vv = 0; vv < 8; ++vv) {
        const int row = bm + (wave << 4) + vv + hi * 8;
        C[(size_t)row * P + col] = acc[vv] + bv;
    }
}

// ---------------------------------------------------------------------------
// Sparse attention core: block = one node, wave = one head, lane = one neighbor.
// ---------------------------------------------------------------------------
__global__ __launch_bounds__(256)
void attn_kernel(const float* __restrict__ q, const float* __restrict__ kk,
                 const float* __restrict__ vv, const int* __restrict__ idx,
                 const unsigned char* __restrict__ mask, float* __restrict__ out,
                 _Float16* __restrict__ out16)
{
    __shared__ float qs[HID];
    const int node = blockIdx.x;
    const int h    = threadIdx.x >> 5;
    const int lane = threadIdx.x & 31;
    qs[threadIdx.x] = q[(size_t)node * HID + threadIdx.x];
    __syncthreads();

    const int  nb  = idx[node * KNBR + lane];
    const bool pad = mask[node * KNBR + lane] != 0;

    const float* kr = kk + (size_t)nb * HID + h * HD;
    float s = 0.f;
    #pragma unroll
    for (int d = 0; d < HD; ++d) s += qs[h * HD + d] * kr[d];
    s *= 0.17677669529663687f;                 // 1/sqrt(32)
    if (pad) s = -__builtin_inff();

    const float m = wave_max(s);
    float e = (pad || m == -__builtin_inff()) ? 0.f : __expf(s - m);
    const float tot = wave_sum(e);
    const float attn = (tot > 0.f) ? (e / tot) : 0.f;

    float vr[HD];
    const float* vrow = vv + (size_t)nb * HID + h * HD;
    #pragma unroll
    for (int d = 0; d < HD; ++d) vr[d] = vrow[d];

    float mine = 0.f;
    #pragma unroll
    for (int d = 0; d < HD; ++d) {
        float r = wave_sum(attn * vr[d]);
        if (lane == d) mine = r;
    }
    out[(size_t)node * HID + h * HD + lane]   = mine;
    out16[(size_t)node * HID + h * HD + lane] = (_Float16)mine;
}

// ---------------------------------------------------------------------------
__global__ __launch_bounds__(256)
void norm_kernel(const float* __restrict__ craw, float* __restrict__ out)
{
    const int n = blockIdx.x * 256 + threadIdx.x;
    float vbuf[16];
    float s = 0.f;
    #pragma unroll
    for (int t = 0; t < 16; ++t) { vbuf[t] = craw[(size_t)n * 16 + t]; s += vbuf[t] * vbuf[t]; }
    const float nm = fmaxf(sqrtf(s), 1e-12f);
    #pragma unroll
    for (int t = 0; t < 16; ++t) out[(size_t)n * 16 + t] = vbuf[t] / nm;
}

__global__ __launch_bounds__(256)
void beta_kernel(const float* __restrict__ t2, const float* __restrict__ w_b2,
                 const float* __restrict__ b_b2, float* __restrict__ out)
{
    const int node = (blockIdx.x * 256 + threadIdx.x) >> 5;
    const int lane = threadIdx.x & 31;
    float s = 0.f;
    #pragma unroll
    for (int t = 0; t < 4; ++t) {
        const int c = lane + t * 32;
        s += t2[(size_t)node * 128 + c] * w_b2[c];
    }
    s = wave_sum(s);
    if (lane == 0) {
        const float z = s + b_b2[0];
        float beta = 1.f / (1.f + __expf(-z));
        beta = fminf(fmaxf(beta, 1e-6f), 1.f - 1e-6f);
        out[node] = beta;
    }
}

// ---------------------------------------------------------------------------
extern "C" void kernel_launch(void* const* d_in, const int* in_sizes, int n_in,
                              void* d_out, int out_size, void* d_ws, size_t ws_size,
                              hipStream_t stream)
{
    const float* x_raw = (const float*)d_in[0];
    const int*   batch = (const int*)d_in[1];
    const float* w_in  = (const float*)d_in[2];
    const float* b_in  = (const float*)d_in[3];
    const float* ln1_g = (const float*)d_in[4];
    const float* ln1_b = (const float*)d_in[5];
    const float* ln2_g = (const float*)d_in[6];
    const float* ln2_b = (const float*)d_in[7];
    const float* wq    = (const float*)d_in[8];
    const float* wk    = (const float*)d_in[9];
    const float* wv    = (const float*)d_in[10];
    const float* wo    = (const float*)d_in[11];
    const float* b_o   = (const float*)d_in[12];
    const float* w_ff1 = (const float*)d_in[13];
    const float* b_ff1 = (const float*)d_in[14];
    const float* w_ff2 = (const float*)d_in[15];
    const float* b_ff2 = (const float*)d_in[16];
    const float* w_l1  = (const float*)d_in[17];
    const float* b_l1  = (const float*)d_in[18];
    const float* w_l2  = (const float*)d_in[19];
    const float* b_l2  = (const float*)d_in[20];
    const float* w_b1  = (const float*)d_in[21];
    const float* b_b1  = (const float*)d_in[22];
    const float* w_b2  = (const float*)d_in[23];
    const float* b_b2  = (const float*)d_in[24];

    // workspace carve-up
    char* w = (char*)d_ws;
    int* idx            = (int*)w;            w += (size_t)NNODES * KNBR * 4;
    unsigned char* mask = (unsigned char*)w;  w += (size_t)NNODES * KNBR;
    float* x   = (float*)w;  w += (size_t)NNODES * HID * 4;
    float* xn  = (float*)w;  w += (size_t)NNODES * HID * 4;
    float* qb  = (float*)w;  w += (size_t)NNODES * HID * 4;
    float* kb  = (float*)w;  w += (size_t)NNODES * HID * 4;
    float* vb  = (float*)w;  w += (size_t)NNODES * HID * 4;
    float* ao  = (float*)w;  w += (size_t)NNODES * HID * 4;
    float* ff  = (float*)w;  w += (size_t)NNODES * 2 * HID * 4;
    _Float16* xn16 = (_Float16*)w;  w += (size_t)NNODES * HID * 2;
    _Float16* ao16 = (_Float16*)w;  w += (size_t)NNODES * HID * 2;
    _Float16* ff16 = (_Float16*)w;  w += (size_t)NNODES * 2 * HID * 2;
    _Float16* x16  = (_Float16*)w;  w += (size_t)NNODES * HID * 2;
    _Float16* qb16 = (_Float16*)w;  w += (size_t)NNODES * HID * 2;
    _Float16* wtq  = (_Float16*)w;  w += (size_t)NLAY * HID * HID * 2;
    _Float16* wtk  = (_Float16*)w;  w += (size_t)NLAY * HID * HID * 2;
    _Float16* wtv  = (_Float16*)w;  w += (size_t)NLAY * HID * HID * 2;
    _Float16* wto  = (_Float16*)w;  w += (size_t)NLAY * HID * HID * 2;
    _Float16* wtf1 = (_Float16*)w;  w += (size_t)NLAY * HID * 2 * HID * 2;
    _Float16* wtf2 = (_Float16*)w;  w += (size_t)NLAY * 2 * HID * HID * 2;
    _Float16* wtl1 = (_Float16*)w;  w += (size_t)HID * HID * 2;
    _Float16* wtb1 = (_Float16*)w;  w += (size_t)HID * 128 * 2;

    float* beta_out   = (float*)d_out;
    float* coords_out = (float*)d_out + NNODES;

    // 0. weight convert+transpose to f16 [P][K]
    for (int l = 0; l < NLAY; ++l) {
        wcvt_kernel<<<HID, 256, 0, stream>>>(wq + (size_t)l * HID * HID, wtq + (size_t)l * HID * HID, HID, HID);
        wcvt_kernel<<<HID, 256, 0, stream>>>(wk + (size_t)l * HID * HID, wtk + (size_t)l * HID * HID, HID, HID);
        wcvt_kernel<<<HID, 256, 0, stream>>>(wv + (size_t)l * HID * HID, wtv + (size_t)l * HID * HID, HID, HID);
        wcvt_kernel<<<HID, 256, 0, stream>>>(wo + (size_t)l * HID * HID, wto + (size_t)l * HID * HID, HID, HID);
        wcvt_kernel<<<2 * HID, 256, 0, stream>>>(w_ff1 + (size_t)l * HID * 2 * HID,
                                                 wtf1 + (size_t)l * HID * 2 * HID, HID, 2 * HID);
        wcvt_kernel<<<HID, 256, 0, stream>>>(w_ff2 + (size_t)l * 2 * HID * HID,
                                             wtf2 + (size_t)l * 2 * HID * HID, 2 * HID, HID);
    }
    wcvt_kernel<<<HID, 256, 0, stream>>>(w_l1, wtl1, HID, HID);
    wcvt_kernel<<<128, 256, 0, stream>>>(w_b1, wtb1, HID, 128);

    // 1. neighbor graph + input projection
    neighbors_kernel<<<NNODES / 256, 256, 0, stream>>>(x_raw, batch, idx, mask);
    inproj_kernel<<<NNODES, 256, 0, stream>>>(x_raw, w_in, b_in, x);

    const dim3 g256(HID / 64, NNODES / 128);      // P=256
    const dim3 g512(2 * HID / 64, NNODES / 128);  // P=512
    const dim3 g128(128 / 64, NNODES / 128);      // P=128
    const dim3 g16 (1,        NNODES / 128);      // P=16 (narrow kernel)

    for (int l = 0; l < NLAY; ++l) {
        ln_kernel<<<NNODES / 8, 256, 0, stream>>>(x, ln1_g + l * HID, ln1_b + l * HID, xn, xn16);
        gemm_wmma_tdm<<<g256, 256, 0, stream>>>(xn16, wtq + (size_t)l * HID * HID,
                                                nullptr, nullptr, qb, nullptr, NNODES, HID, HID, 0);
        gemm_wmma_tdm<<<g256, 256, 0, stream>>>(xn16, wtk + (size_t)l * HID * HID,
                                                nullptr, nullptr, kb, nullptr, NNODES, HID, HID, 0);
        gemm_wmma_tdm<<<g256, 256, 0, stream>>>(xn16, wtv + (size_t)l * HID * HID,
                                                nullptr, nullptr, vb, nullptr, NNODES, HID, HID, 0);
        attn_kernel<<<NNODES, 256, 0, stream>>>(qb, kb, vb, idx, mask, ao, ao16);
        gemm_wmma_tdm<<<g256, 256, 0, stream>>>(ao16, wto + (size_t)l * HID * HID,
                                                b_o + l * HID, x, x, x16, NNODES, HID, HID, 0);

        ln_kernel<<<NNODES / 8, 256, 0, stream>>>(x, ln2_g + l * HID, ln2_b + l * HID, xn, xn16);
        gemm_wmma_tdm<<<g512, 256, 0, stream>>>(xn16, wtf1 + (size_t)l * HID * 2 * HID,
                                                b_ff1 + l * 2 * HID, nullptr, ff, ff16,
                                                NNODES, HID, 2 * HID, 1);
        gemm_wmma_tdm<<<g256, 256, 0, stream>>>(ff16, wtf2 + (size_t)l * 2 * HID * HID,
                                                b_ff2 + l * HID, x, x, x16,
                                                NNODES, 2 * HID, HID, 0);
    }

    // coords head
    gemm_wmma_tdm<<<g256, 256, 0, stream>>>(x16, wtl1, b_l1, nullptr, qb, qb16,
                                            NNODES, HID, HID, 1);
    gemm_wmma16<<<g16, 256, 0, stream>>>(qb, w_l2, b_l2, ao, NNODES, HID, 16);
    norm_kernel<<<NNODES / 256, 256, 0, stream>>>(ao, coords_out);

    // beta head
    gemm_wmma_tdm<<<g128, 256, 0, stream>>>(x16, wtb1, b_b1, nullptr, kb, nullptr,
                                            NNODES, HID, 128, 1);
    beta_kernel<<<NNODES / 8, 256, 0, stream>>>(kb, w_b2, b_b2, beta_out);
}
